// DFlashAttention_62122406969388
// MI455X (gfx1250) — matrix operand order repaired and checked by reading, compiled, not run on hardware
//
#include <hip/hip_runtime.h>
#include <hip/hip_bf16.h>

// ---------------------------------------------------------------------------
// DFlashAttention for MI455X (gfx1250): bf16 WMMA pipeline.
//  - all LDS fragment traffic is ds_load_b128 (contiguous per-lane runs)
//  - attention K/Q tiles staged by the Tensor Data Mover (tensor_load_to_lds
//    + s_wait_tensorcnt) when the toolchain exposes the builtin
//  - softmax runs in base-2 with 1/sqrt(d)*log2(e) folded into Q
// ---------------------------------------------------------------------------

typedef __bf16 bf16_t;
typedef __attribute__((ext_vector_type(16))) __bf16 v16bf;
typedef __attribute__((ext_vector_type(8)))  __bf16 v8bf;
typedef __attribute__((ext_vector_type(8)))  float  v8f;

#define DMODEL 2048
#define HD     64
#define NH     32
#define NKVH   8
#define CTX    4096
#define QL     64
#define KVLEN  4160
#define NB     4
#define KC     32
#define LDA    40     // A tile stride (80B: 16B-aligned, conflict-free)
#define LDBT   40     // transposed B tile stride
#define LDT    72     // attn tile stride (144B: 16B-aligned)

// 0.125 (1/sqrt(64)) * log2(e): softmax done in base 2
#define QSCALE 0.18033688011112042f

// ---- TDM (Tensor Data Mover) support --------------------------------------

#if __has_builtin(__builtin_amdgcn_tensor_load_to_lds) && \
    __has_builtin(__builtin_amdgcn_s_wait_tensorcnt)
#define ATHENA_HAS_TDM 1
#endif

#ifdef ATHENA_HAS_TDM
typedef __attribute__((ext_vector_type(4))) unsigned tdm_v4u;
typedef __attribute__((ext_vector_type(8))) int      tdm_v8i;
typedef __attribute__((ext_vector_type(4))) int      tdm_v4i;

// 2-D bf16 tile load: tile_w x tile_h elements, row stride in elements,
// LDS padding of pad_amount-code DWORDs every pad_interval-code DWORDs.
__device__ __forceinline__ void tdm_load_2d(const void* gsrc, const void* lds_dst,
                                            int tile_w, int tile_h,
                                            int row_stride_elems,
                                            int pad_interval_code,
                                            int pad_amount_code) {
  unsigned long long ga = (unsigned long long)(uintptr_t)gsrc;
  unsigned lo = (unsigned)(uintptr_t)lds_dst;   // generic LDS addr[31:0] = offset
  tdm_v4u g0;
  g0[0] = 1u;                                   // count=1 (valid), user mode
  g0[1] = lo;                                   // lds_addr
  g0[2] = (unsigned)ga;                         // global_addr[31:0]
  g0[3] = (unsigned)((ga >> 32) & 0x01ffffffu) | (2u << 30);  // [56:32] | type=2
  tdm_v8i g1;
  g1[0] = (1 << 16)                             // data_size: 2 bytes (bf16)
        | (1 << 20)                             // pad_enable
        | (pad_interval_code << 22)
        | (pad_amount_code << 25);
  g1[1] = (tile_w & 0xffff) << 16;              // tensor_dim0 lo16 (bits 63:48)
  g1[2] = ((tile_w >> 16) & 0xffff) | ((tile_h & 0xffff) << 16);  // dim0 hi|dim1 lo
  g1[3] = ((tile_h >> 16) & 0xffff) | ((tile_w & 0xffff) << 16);  // dim1 hi|tile_dim0
  g1[4] = (tile_h & 0xffff);                    // tile_dim1 (tile_dim2=0)
  g1[5] = row_stride_elems;                     // tensor_dim0_stride lo32
  g1[6] = 0;
  g1[7] = 0;
  tdm_v4i z = {};
#if __has_include(<hip/amd_detail/amd_gfx1250_TDM.h>)
  tdm_v8i z8 = {};
  __builtin_amdgcn_tensor_load_to_lds(g0, g1, z, z, z8, 0);
#else
  __builtin_amdgcn_tensor_load_to_lds(g0, g1, z, z, 0);
#endif
}
#endif

// ---- WMMA helpers ----------------------------------------------------------

__device__ __forceinline__ v8f wmma_bf16(v16bf a, v16bf b, v8f c) {
  return __builtin_amdgcn_wmma_f32_16x16x32_bf16(
      false, a, false, b, (short)0, c, false, false);
}

__device__ __forceinline__ v16bf load_a_frag(const bf16_t* a, int lda,
                                             int mbase, int lane) {
  const int half = lane >> 4;
  const bf16_t* row = a + (mbase + (lane & 15)) * lda + half * 8;
  v8bf lo = *(const v8bf*)(row);
  v8bf hi = *(const v8bf*)(row + 16);
  return __builtin_shufflevector(lo, hi, 0, 1, 2, 3, 4, 5, 6, 7,
                                 8, 9, 10, 11, 12, 13, 14, 15);
}

__device__ __forceinline__ v16bf load_b_frag(const bf16_t* bt, int ldbt,
                                             int nbase, int lane) {
  const int half = lane >> 4;
  const bf16_t* row = bt + (nbase + (lane & 15)) * ldbt + half * 8;
  v8bf lo = *(const v8bf*)(row);
  v8bf hi = *(const v8bf*)(row + 16);
  return __builtin_shufflevector(lo, hi, 0, 1, 2, 3, 4, 5, 6, 7,
                                 8, 9, 10, 11, 12, 13, 14, 15);
}

__device__ __forceinline__ unsigned pack2(float a, float b) {
  union { bf16_t h[2]; unsigned u; } p;
  p.h[0] = (bf16_t)a; p.h[1] = (bf16_t)b;
  return p.u;
}

__device__ __forceinline__ v8bf cvt8(const float4 a, const float4 b) {
  v8bf r;
  r[0] = (bf16_t)a.x; r[1] = (bf16_t)a.y; r[2] = (bf16_t)a.z; r[3] = (bf16_t)a.w;
  r[4] = (bf16_t)b.x; r[5] = (bf16_t)b.y; r[6] = (bf16_t)b.z; r[7] = (bf16_t)b.w;
  return r;
}

// ---- Kernel 1: K/V projection + rmsnorm + rope -----------------------------

__global__ __launch_bounds__(256)
void kv_proj_kernel(const float* __restrict__ hs, const float* __restrict__ th,
                    const float* __restrict__ Wk, const float* __restrict__ Wv,
                    const float* __restrict__ cosT, const float* __restrict__ sinT,
                    const float* __restrict__ knw,
                    bf16_t* __restrict__ Kout, bf16_t* __restrict__ Vout) {
  __shared__ __align__(16) bf16_t As[64 * LDA];
  __shared__ __align__(16) bf16_t Bkt[HD * LDBT];
  __shared__ __align__(16) bf16_t Bvt[HD * LDBT];

  const int ptile = blockIdx.x;
  const int kvh   = blockIdx.y;
  const int batch = blockIdx.z;
  const int tid  = threadIdx.x;
  const int lane = tid & 31;
  const int wave = tid >> 5;
  const int isV  = wave >> 2;
  const int mrow = (wave & 3) * 16;
  const int pos0 = ptile * 64;

  const int ar  = tid >> 2;
  const int ac0 = (tid & 3) * 8;
  const int ap  = pos0 + ar;
  const float* asrc = (ap < CTX)
      ? (th + ((size_t)batch * CTX + ap) * DMODEL + ac0)
      : (hs + ((size_t)batch * QL + (ap - CTX)) * DMODEL + ac0);
  const int k0 = (tid >> 4) * 2;
  const int n0 = (tid & 15) * 4;
  const float* wk0 = Wk + (size_t)k0 * (NKVH * HD) + kvh * HD + n0;
  const float* wv0 = Wv + (size_t)k0 * (NKVH * HD) + kvh * HD + n0;

  v8f acc[4] = {};

#pragma unroll 1
  for (int kb = 0; kb < DMODEL; kb += KC) {
    __syncthreads();
    {
      float4 f0 = *(const float4*)(asrc + kb);
      float4 f1 = *(const float4*)(asrc + kb + 4);
      *(v8bf*)&As[ar * LDA + ac0] = cvt8(f0, f1);
      if (kb + KC < DMODEL) __builtin_prefetch(asrc + kb + KC, 0, 1);
    }
    {
      const size_t koff = (size_t)kb * (NKVH * HD);
      float4 a0 = *(const float4*)(wk0 + koff);
      float4 a1 = *(const float4*)(wk0 + koff + NKVH * HD);
      float4 b0 = *(const float4*)(wv0 + koff);
      float4 b1 = *(const float4*)(wv0 + koff + NKVH * HD);
      *(unsigned*)&Bkt[(n0 + 0) * LDBT + k0] = pack2(a0.x, a1.x);
      *(unsigned*)&Bkt[(n0 + 1) * LDBT + k0] = pack2(a0.y, a1.y);
      *(unsigned*)&Bkt[(n0 + 2) * LDBT + k0] = pack2(a0.z, a1.z);
      *(unsigned*)&Bkt[(n0 + 3) * LDBT + k0] = pack2(a0.w, a1.w);
      *(unsigned*)&Bvt[(n0 + 0) * LDBT + k0] = pack2(b0.x, b1.x);
      *(unsigned*)&Bvt[(n0 + 1) * LDBT + k0] = pack2(b0.y, b1.y);
      *(unsigned*)&Bvt[(n0 + 2) * LDBT + k0] = pack2(b0.z, b1.z);
      *(unsigned*)&Bvt[(n0 + 3) * LDBT + k0] = pack2(b0.w, b1.w);
    }
    __syncthreads();
    const bf16_t* Bt = isV ? Bvt : Bkt;
    v16bf af = load_a_frag(As, LDA, mrow, lane);
    v16bf bfv[4];
#pragma unroll
    for (int j = 0; j < 4; ++j) bfv[j] = load_b_frag(Bt, LDBT, j * 16, lane);
#pragma unroll
    for (int j = 0; j < 4; ++j) acc[j] = wmma_bf16(af, bfv[j], acc[j]);
  }

  const int half = lane >> 4;
  const int n15  = lane & 15;
  bf16_t* base = (isV ? Vout : Kout) +
                 (((size_t)batch * NKVH + kvh) * KVLEN) * HD;

  if (isV) {
#pragma unroll
    for (int i = 0; i < 8; ++i) {
      int p = pos0 + mrow + i + half * 8;
#pragma unroll
      for (int j = 0; j < 4; ++j)
        base[(size_t)p * HD + j * 16 + n15] = (bf16_t)acc[j][i];
    }
  } else {
#pragma unroll
    for (int i = 0; i < 8; ++i) {
      float ss = 0.f;
#pragma unroll
      for (int j = 0; j < 4; ++j) ss += acc[j][i] * acc[j][i];
      ss += __shfl_xor(ss, 1); ss += __shfl_xor(ss, 2);
      ss += __shfl_xor(ss, 4); ss += __shfl_xor(ss, 8);
      float inv = rsqrtf(ss * (1.0f / HD) + 1e-6f);
      int p = pos0 + mrow + i + half * 8;
      const float* cp = cosT + ((size_t)batch * KVLEN + p) * HD;
      const float* sp = sinT + ((size_t)batch * KVLEN + p) * HD;
      float x[4];
#pragma unroll
      for (int j = 0; j < 4; ++j)
        x[j] = acc[j][i] * inv * knw[j * 16 + n15];
#pragma unroll
      for (int j = 0; j < 4; ++j) {
        int n = j * 16 + n15;
        float rot = (j < 2) ? -x[j + 2] : x[j - 2];
        base[(size_t)p * HD + n] = (bf16_t)(x[j] * cp[n] + rot * sp[n]);
      }
    }
  }
}

// ---- Kernel 2: Q projection + rmsnorm + rope (scaled for softmax) ----------

__global__ __launch_bounds__(128)
void q_proj_kernel(const float* __restrict__ hs, const float* __restrict__ Wq,
                   const float* __restrict__ cosT, const float* __restrict__ sinT,
                   const float* __restrict__ qnw, bf16_t* __restrict__ Qout) {
  __shared__ __align__(16) bf16_t As[64 * LDA];
  __shared__ __align__(16) bf16_t Bt[HD * LDBT];

  const int h     = blockIdx.x;
  const int batch = blockIdx.y;
  const int tid  = threadIdx.x;
  const int lane = tid & 31;
  const int wave = tid >> 5;
  const int mrow = wave * 16;

  const int ar  = tid >> 1;
  const int ac0 = (tid & 1) * 16;
  const float* asrc = hs + ((size_t)batch * QL + ar) * DMODEL + ac0;
  const int k0 = (tid >> 3) * 2;
  const int n0 = (tid & 7) * 8;
  const float* w0 = Wq + (size_t)k0 * (NH * HD) + h * HD + n0;

  v8f acc[4] = {};

#pragma unroll 1
  for (int kb = 0; kb < DMODEL; kb += KC) {
    __syncthreads();
    {
      float4 f0 = *(const float4*)(asrc + kb);
      float4 f1 = *(const float4*)(asrc + kb + 4);
      float4 f2 = *(const float4*)(asrc + kb + 8);
      float4 f3 = *(const float4*)(asrc + kb + 12);
      *(v8bf*)&As[ar * LDA + ac0]     = cvt8(f0, f1);
      *(v8bf*)&As[ar * LDA + ac0 + 8] = cvt8(f2, f3);
      if (kb + KC < DMODEL) __builtin_prefetch(asrc + kb + KC, 0, 1);
    }
    {
      const size_t koff = (size_t)kb * (NH * HD);
      float4 a0 = *(const float4*)(w0 + koff);
      float4 a1 = *(const float4*)(w0 + koff + 4);
      float4 b0 = *(const float4*)(w0 + koff + NH * HD);
      float4 b1 = *(const float4*)(w0 + koff + NH * HD + 4);
      *(unsigned*)&Bt[(n0 + 0) * LDBT + k0] = pack2(a0.x, b0.x);
      *(unsigned*)&Bt[(n0 + 1) * LDBT + k0] = pack2(a0.y, b0.y);
      *(unsigned*)&Bt[(n0 + 2) * LDBT + k0] = pack2(a0.z, b0.z);
      *(unsigned*)&Bt[(n0 + 3) * LDBT + k0] = pack2(a0.w, b0.w);
      *(unsigned*)&Bt[(n0 + 4) * LDBT + k0] = pack2(a1.x, b1.x);
      *(unsigned*)&Bt[(n0 + 5) * LDBT + k0] = pack2(a1.y, b1.y);
      *(unsigned*)&Bt[(n0 + 6) * LDBT + k0] = pack2(a1.z, b1.z);
      *(unsigned*)&Bt[(n0 + 7) * LDBT + k0] = pack2(a1.w, b1.w);
    }
    __syncthreads();
    v16bf af = load_a_frag(As, LDA, mrow, lane);
    v16bf bfv[4];
#pragma unroll
    for (int j = 0; j < 4; ++j) bfv[j] = load_b_frag(Bt, LDBT, j * 16, lane);
#pragma unroll
    for (int j = 0; j < 4; ++j) acc[j] = wmma_bf16(af, bfv[j], acc[j]);
  }

  const int half = lane >> 4;
  const int n15  = lane & 15;
  bf16_t* qb = Qout + (((size_t)batch * NH + h) * QL) * HD;

#pragma unroll
  for (int i = 0; i < 8; ++i) {
    float ss = 0.f;
#pragma unroll
    for (int j = 0; j < 4; ++j) ss += acc[j][i] * acc[j][i];
    ss += __shfl_xor(ss, 1); ss += __shfl_xor(ss, 2);
    ss += __shfl_xor(ss, 4); ss += __shfl_xor(ss, 8);
    float inv = rsqrtf(ss * (1.0f / HD) + 1e-6f);
    int qrow = mrow + i + half * 8;
    int p = CTX + qrow;
    const float* cp = cosT + ((size_t)batch * KVLEN + p) * HD;
    const float* sp = sinT + ((size_t)batch * KVLEN + p) * HD;
    float x[4];
#pragma unroll
    for (int j = 0; j < 4; ++j)
      x[j] = acc[j][i] * inv * qnw[j * 16 + n15];
#pragma unroll
    for (int j = 0; j < 4; ++j) {
      int n = j * 16 + n15;
      float rot = (j < 2) ? -x[j + 2] : x[j - 2];
      // fold 1/sqrt(d) * log2(e) into Q so attention softmax runs in base 2
      qb[(size_t)qrow * HD + n] =
          (bf16_t)((x[j] * cp[n] + rot * sp[n]) * QSCALE);
    }
  }
}

// ---- Kernel 3: flash attention ---------------------------------------------

__global__ __launch_bounds__(128)
void attn_kernel(const bf16_t* __restrict__ Q, const bf16_t* __restrict__ Kc,
                 const bf16_t* __restrict__ Vc, bf16_t* __restrict__ AO) {
  __shared__ __align__(16) bf16_t Kt[64 * LDT];       // natural [pos][hd]
  __shared__ __align__(16) bf16_t Vt[HD * LDT];       // transposed [hd][pos]
  __shared__ __align__(16) bf16_t Pt[4 * 16 * LDT];   // per-wave P / Q staging

  const int h     = blockIdx.x;
  const int batch = blockIdx.y;
  const int kvh   = h / (NH / NKVH);
  const int tid  = threadIdx.x;
  const int lane = tid & 31;
  const int wave = tid >> 5;
  const int half = lane >> 4;
  const int n15  = lane & 15;

  // stage Q (64x64) into Pt, build persistent A fragments
  const bf16_t* Qb = Q + (((size_t)batch * NH + h) * QL) * HD;
#ifdef ATHENA_HAS_TDM
  if (wave == 0) {
    tdm_load_2d(Qb, Pt, 64, 64, HD, /*interval 32DW*/4, /*amount 4DW*/3);
    __builtin_amdgcn_s_wait_tensorcnt(0);
  }
#else
  {
    int r  = tid >> 1;
    int c0 = (tid & 1) * 32;
    const v8bf* src = (const v8bf*)(Qb + (size_t)r * HD + c0);
    v8bf* dst = (v8bf*)&Pt[r * LDT + c0];
    dst[0] = src[0]; dst[1] = src[1]; dst[2] = src[2]; dst[3] = src[3];
  }
#endif
  __syncthreads();
  v16bf qf0 = load_a_frag(Pt,      LDT, wave * 16, lane);
  v16bf qf1 = load_a_frag(Pt + 32, LDT, wave * 16, lane);

  const bf16_t* Kb = Kc + (((size_t)batch * NKVH + kvh) * KVLEN) * HD;
  const bf16_t* Vb = Vc + (((size_t)batch * NKVH + kvh) * KVLEN) * HD;

  const int vp0 = (tid >> 2) * 2;     // V: position pair
  const int vc0 = (tid & 3) * 16;     // V: hd column group
#ifndef ATHENA_HAS_TDM
  const int kr  = tid >> 1;
  const int kc0 = (tid & 1) * 32;
#endif

  v8f o[4] = {};
  float m_r[8], l_r[8];
#pragma unroll
  for (int i = 0; i < 8; ++i) { m_r[i] = -1e30f; l_r[i] = 0.f; }

#pragma unroll 1
  for (int t = 0; t < KVLEN / 64; ++t) {
    __syncthreads();
#ifdef ATHENA_HAS_TDM
    if (wave == 0)                     // DMA the K tile while waves stage V
      tdm_load_2d(Kb + (size_t)t * 64 * HD, Kt, 64, 64, HD, 4, 3);
#else
    {
      const v8bf* src = (const v8bf*)(Kb + ((size_t)(t * 64 + kr)) * HD + kc0);
      v8bf* dst = (v8bf*)&Kt[kr * LDT + kc0];
      dst[0] = src[0]; dst[1] = src[1]; dst[2] = src[2]; dst[3] = src[3];
    }
#endif
    {  // V tile transposed: row-pair packed b32 stores
      const bf16_t* v0 = Vb + ((size_t)(t * 64 + vp0)) * HD + vc0;
      const bf16_t* v1 = v0 + HD;
#pragma unroll
      for (int c = 0; c < 16; ++c)
        *(unsigned*)&Vt[(vc0 + c) * LDT + vp0] = pack2((float)v0[c], (float)v1[c]);
    }
#ifdef ATHENA_HAS_TDM
    if (wave == 0) __builtin_amdgcn_s_wait_tensorcnt(0);
#endif
    __syncthreads();

    // S = Q @ K^T : preload all 8 B-frags, then 8 WMMAs
    v16bf kb0[4], kb1[4];
#pragma unroll
    for (int j = 0; j < 4; ++j) kb0[j] = load_b_frag(Kt,      LDT, j * 16, lane);
#pragma unroll
    for (int j = 0; j < 4; ++j) kb1[j] = load_b_frag(Kt + 32, LDT, j * 16, lane);
    v8f s[4] = {};
#pragma unroll
    for (int j = 0; j < 4; ++j) s[j] = wmma_bf16(qf0, kb0[j], s[j]);
#pragma unroll
    for (int j = 0; j < 4; ++j) s[j] = wmma_bf16(qf1, kb1[j], s[j]);

    // online softmax in base 2 (scale already folded into Q)
#pragma unroll
    for (int i = 0; i < 8; ++i) {
      float mx = -1e30f;
#pragma unroll
      for (int j = 0; j < 4; ++j) mx = fmaxf(mx, s[j][i]);
      mx = fmaxf(mx, __shfl_xor(mx, 1));
      mx = fmaxf(mx, __shfl_xor(mx, 2));
      mx = fmaxf(mx, __shfl_xor(mx, 4));
      mx = fmaxf(mx, __shfl_xor(mx, 8));
      float mnew  = fmaxf(m_r[i], mx);
      float alpha = exp2f(m_r[i] - mnew);
      m_r[i] = mnew;
      float rs = 0.f;
#pragma unroll
      for (int j = 0; j < 4; ++j) {
        float pv = exp2f(s[j][i] - mnew);
        s[j][i] = pv;
        rs += pv;
      }
      rs += __shfl_xor(rs, 1); rs += __shfl_xor(rs, 2);
      rs += __shfl_xor(rs, 4); rs += __shfl_xor(rs, 8);
      l_r[i] = l_r[i] * alpha + rs;
#pragma unroll
      for (int j = 0; j < 4; ++j) o[j][i] *= alpha;
    }

    // P (C layout) -> per-wave LDS slab
    bf16_t* Pw = Pt + wave * 16 * LDT;
#pragma unroll
    for (int i = 0; i < 8; ++i) {
      int r = i + half * 8;
#pragma unroll
      for (int j = 0; j < 4; ++j)
        Pw[r * LDT + j * 16 + n15] = (bf16_t)s[j][i];
    }
    __syncthreads();

    // O += P @ V
#pragma unroll
    for (int ks = 0; ks < 2; ++ks) {
      v16bf pf = load_a_frag(Pw + ks * 32, LDT, 0, lane);
      v16bf vb[4];
#pragma unroll
      for (int j = 0; j < 4; ++j)
        vb[j] = load_b_frag(Vt + ks * 32, LDT, j * 16, lane);
#pragma unroll
      for (int j = 0; j < 4; ++j) o[j] = wmma_bf16(pf, vb[j], o[j]);
    }
  }

  bf16_t* Ao = AO + ((size_t)batch * QL) * (NH * HD) + h * HD;
#pragma unroll
  for (int i = 0; i < 8; ++i) {
    int qrow = wave * 16 + i + half * 8;
    float invl = 1.0f / l_r[i];
#pragma unroll
    for (int j = 0; j < 4; ++j)
      Ao[(size_t)qrow * (NH * HD) + j * 16 + n15] = (bf16_t)(o[j][i] * invl);
  }
}

// ---- Kernel 4: output projection -------------------------------------------

__global__ __launch_bounds__(128)
void out_proj_kernel(const bf16_t* __restrict__ A, const float* __restrict__ Wo,
                     float* __restrict__ Out) {
  __shared__ __align__(16) bf16_t As[64 * LDA];
  __shared__ __align__(16) bf16_t Bt[HD * LDBT];

  const int nt = blockIdx.x;
  const int mt = blockIdx.y;
  const int tid  = threadIdx.x;
  const int lane = tid & 31;
  const int wave = tid >> 5;
  const int mrow = wave * 16;

  const int ar  = tid >> 1;
  const int ac0 = (tid & 1) * 16;
  const bf16_t* asrc = A + ((size_t)(mt * 64 + ar)) * DMODEL + ac0;
  const int k0 = (tid >> 3) * 2;
  const int n0 = (tid & 7) * 8;
  const float* w0 = Wo + (size_t)k0 * DMODEL + nt * 64 + n0;

  v8f acc[4] = {};

#pragma unroll 1
  for (int kb = 0; kb < DMODEL; kb += KC) {
    __syncthreads();
    {
      const v8bf* src = (const v8bf*)(asrc + kb);
      *(v8bf*)&As[ar * LDA + ac0]     = src[0];
      *(v8bf*)&As[ar * LDA + ac0 + 8] = src[1];
      if (kb + KC < DMODEL) __builtin_prefetch(asrc + kb + KC, 0, 1);
    }
    {
      const size_t koff = (size_t)kb * DMODEL;
      float4 a0 = *(const float4*)(w0 + koff);
      float4 a1 = *(const float4*)(w0 + koff + 4);
      float4 b0 = *(const float4*)(w0 + koff + DMODEL);
      float4 b1 = *(const float4*)(w0 + koff + DMODEL + 4);
      *(unsigned*)&Bt[(n0 + 0) * LDBT + k0] = pack2(a0.x, b0.x);
      *(unsigned*)&Bt[(n0 + 1) * LDBT + k0] = pack2(a0.y, b0.y);
      *(unsigned*)&Bt[(n0 + 2) * LDBT + k0] = pack2(a0.z, b0.z);
      *(unsigned*)&Bt[(n0 + 3) * LDBT + k0] = pack2(a0.w, b0.w);
      *(unsigned*)&Bt[(n0 + 4) * LDBT + k0] = pack2(a1.x, b1.x);
      *(unsigned*)&Bt[(n0 + 5) * LDBT + k0] = pack2(a1.y, b1.y);
      *(unsigned*)&Bt[(n0 + 6) * LDBT + k0] = pack2(a1.z, b1.z);
      *(unsigned*)&Bt[(n0 + 7) * LDBT + k0] = pack2(a1.w, b1.w);
    }
    __syncthreads();
    v16bf af = load_a_frag(As, LDA, mrow, lane);
    v16bf bfv[4];
#pragma unroll
    for (int j = 0; j < 4; ++j) bfv[j] = load_b_frag(Bt, LDBT, j * 16, lane);
#pragma unroll
    for (int j = 0; j < 4; ++j) acc[j] = wmma_bf16(af, bfv[j], acc[j]);
  }

  const int half = lane >> 4;
  const int n15  = lane & 15;
#pragma unroll
  for (int i = 0; i < 8; ++i) {
    int row = mt * 64 + mrow + i + half * 8;
#pragma unroll
    for (int j = 0; j < 4; ++j)
      Out[(size_t)row * DMODEL + nt * 64 + j * 16 + n15] = acc[j][i];
  }
}

// ---- Host launcher ---------------------------------------------------------

extern "C" void kernel_launch(void* const* d_in, const int* in_sizes, int n_in,
                              void* d_out, int out_size, void* d_ws, size_t ws_size,
                              hipStream_t stream) {
  (void)in_sizes; (void)n_in; (void)out_size; (void)ws_size;

  const float* hs   = (const float*)d_in[0];
  const float* th   = (const float*)d_in[1];
  const float* cosT = (const float*)d_in[2];
  const float* sinT = (const float*)d_in[3];
  const float* Wq   = (const float*)d_in[4];
  const float* Wk   = (const float*)d_in[5];
  const float* Wv   = (const float*)d_in[6];
  const float* Wo   = (const float*)d_in[7];
  const float* qnw  = (const float*)d_in[8];
  const float* knw  = (const float*)d_in[9];
  float* Out = (float*)d_out;

  bf16_t* Qws = (bf16_t*)d_ws;
  bf16_t* Kws = Qws + (size_t)NB * NH * QL * HD;
  bf16_t* Vws = Kws + (size_t)NB * NKVH * KVLEN * HD;
  bf16_t* Aws = Vws + (size_t)NB * NKVH * KVLEN * HD;

  kv_proj_kernel<<<dim3(KVLEN / 64, NKVH, NB), 256, 0, stream>>>(
      hs, th, Wk, Wv, cosT, sinT, knw, Kws, Vws);
  q_proj_kernel<<<dim3(NH, NB), 128, 0, stream>>>(
      hs, Wq, cosT, sinT, qnw, Qws);
  attn_kernel<<<dim3(NH, NB), 128, 0, stream>>>(
      Qws, Kws, Vws, Aws);
  out_proj_kernel<<<dim3(DMODEL / 64, (NB * QL) / 64), 128, 0, stream>>>(
      Aws, Wo, Out);
}
